// WinningRateReversed_69355131896602
// MI455X (gfx1250) — compile-verified, matrix operand order checked by reading, not run on hardware
//
#include <hip/hip_runtime.h>

typedef __attribute__((ext_vector_type(8))) int   v8i;
typedef __attribute__((ext_vector_type(4))) float v4f;

// Streaming win-rate counter.
// Per wave tile: 8 j-steps x 32 lanes x (float4 x + float4 y) = 1024 element
// pairs -> 1024 compare bytes -> one V_WMMA_I32_16X16X64_IU8 against an
// all-ones B matrix. Sum of all 256 D elements == 16 * (# of true compares).
__global__ __launch_bounds__(256) void winrate_count_kernel(
    const v4f* __restrict__ x4,
    const v4f* __restrict__ y4,
    unsigned int* __restrict__ g_count,
    unsigned int n4)   // number of float4 pairs (< 2^31)
{
    const unsigned lane = threadIdx.x & 31u;

    // Scalarize wave id / loop bounds: loop control in SGPRs, EXEC untouched.
    const unsigned waveId =
        __builtin_amdgcn_readfirstlane(
            (blockIdx.x * blockDim.x + threadIdx.x) >> 5);
    const unsigned nWaves = (gridDim.x * blockDim.x) >> 5;
    const unsigned stride = nWaves * 256u;   // float4s per grid pass

    v8i ones;
    #pragma unroll
    for (int i = 0; i < 8; ++i) ones[i] = 0x01010101;

    v8i acc;
    #pragma unroll
    for (int i = 0; i < 8; ++i) acc[i] = 0;

    unsigned base = waveId * 256u;           // SGPR

    // ---- Fast path: whole 256-float4 tile in bounds (scalar condition). ----
    // 16 unguarded global_load_b128 issued back-to-back -> deep MLP.
    for (; base + 256u <= n4; base += stride) {
        v4f xv[8], yv[8];
        #pragma unroll
        for (int j = 0; j < 8; ++j) {
            const unsigned idx = base + (unsigned)j * 32u + lane;
            xv[j] = __builtin_nontemporal_load(&x4[idx]);
            yv[j] = __builtin_nontemporal_load(&y4[idx]);
        }
        v8i a;
        #pragma unroll
        for (int j = 0; j < 8; ++j) {
            unsigned w;
            w  = (xv[j].x < yv[j].x) ? 0x00000001u : 0u;
            w |= (xv[j].y < yv[j].y) ? 0x00000100u : 0u;
            w |= (xv[j].z < yv[j].z) ? 0x00010000u : 0u;
            w |= (xv[j].w < yv[j].w) ? 0x01000000u : 0u;
            a[j] = (int)w;
        }
        // (sgn_a, A, sgn_b, B, C, reuse_a, reuse_b)
        acc = __builtin_amdgcn_wmma_i32_16x16x64_iu8(
            false, a, false, ones, acc, false, false);
    }

    // ---- Tail: at most one partial tile per wave (stride >= 256). ----
    if (base < n4) {
        v8i a;
        #pragma unroll
        for (int j = 0; j < 8; ++j) {
            const unsigned idx = base + (unsigned)j * 32u + lane;
            unsigned w = 0u;
            if (idx < n4) {
                const v4f xv = __builtin_nontemporal_load(&x4[idx]);
                const v4f yv = __builtin_nontemporal_load(&y4[idx]);
                w  = (xv.x < yv.x) ? 0x00000001u : 0u;
                w |= (xv.y < yv.y) ? 0x00000100u : 0u;
                w |= (xv.z < yv.z) ? 0x00010000u : 0u;
                w |= (xv.w < yv.w) ? 0x01000000u : 0u;
            }
            a[j] = (int)w;
        }
        // EXEC restored (guards only wrap the loads) -> legal WMMA.
        acc = __builtin_amdgcn_wmma_i32_16x16x64_iu8(
            false, a, false, ones, acc, false, false);
    }

    // Per-lane partial over the 8 accumulator VGPRs.
    int s = 0;
    #pragma unroll
    for (int i = 0; i < 8; ++i) s += acc[i];

    // wave32 butterfly reduction.
    #pragma unroll
    for (int off = 16; off > 0; off >>= 1)
        s += __shfl_xor(s, off, 32);

    if (lane == 0)
        atomicAdd(g_count, (unsigned int)s);  // holds 16 * true_count
}

__global__ void winrate_finalize_kernel(const unsigned int* __restrict__ g_count,
                                        float* __restrict__ out,
                                        long long n_elems)
{
    if (threadIdx.x == 0 && blockIdx.x == 0) {
        const unsigned int c16 = *g_count;
        const float cnt = (float)(c16 / 16u);   // exact: every D column identical
        out[0] = 1.0f - cnt / (float)n_elems;
    }
}

extern "C" void kernel_launch(void* const* d_in, const int* in_sizes, int n_in,
                              void* d_out, int out_size, void* d_ws, size_t ws_size,
                              hipStream_t stream)
{
    const float* x = (const float*)d_in[0];
    const float* y = (const float*)d_in[1];
    float* out = (float*)d_out;

    const long long n  = (long long)in_sizes[0];  // 64 * 1e6, divisible by 4
    const unsigned  n4 = (unsigned)(n >> 2);      // 16e6 < 2^31

    unsigned int* counter = (unsigned int*)d_ws;

    // Zero the accumulator (d_ws is poisoned; memsetAsync is capture-safe).
    hipMemsetAsync(counter, 0, sizeof(unsigned int), stream);

    // 2048 blocks x 256 threads = 16K waves: enough MLP to saturate 23.3 TB/s.
    const int threads = 256;
    const int blocks  = 2048;
    winrate_count_kernel<<<blocks, threads, 0, stream>>>(
        (const v4f*)x, (const v4f*)y, counter, n4);

    winrate_finalize_kernel<<<1, 32, 0, stream>>>(counter, out, n);
}
